// Merger_54425825575059
// MI455X (gfx1250) — matrix operand order repaired and checked by reading, not compiled, not run on hardware
//
#include <hip/hip_runtime.h>

#define B_  4
#define C_  512
#define H_  128
#define W_  128
#define HW_ (H_*W_)
#define N_  512   // MERGER

typedef __attribute__((ext_vector_type(16))) __bf16 v16bf;
typedef __attribute__((ext_vector_type(8)))  float  v8f;

union FragAB { v16bf v; unsigned int u[8]; };

__device__ __forceinline__ unsigned short f2bf(float f) {
  unsigned int u = __builtin_bit_cast(unsigned int, f);
  u += 0x7fffu + ((u >> 16) & 1u);   // round-to-nearest-even
  return (unsigned short)(u >> 16);
}

// LDS strides in 32-bit units (each uint = 2 bf16 along k)
#define LDA32 17   // A: 64 rows x 16 k-pairs, stride 17 uints (conflict-free)
#define LDB32 20   // B: 512 rows x 16 k-pairs, stride 20 uints (80B rows, 16B-aligned)

__device__ __forceinline__ unsigned lds_off(const void* p) {
  // generic (shared) pointer -> 32-bit LDS byte offset
  return (unsigned)(unsigned long long)
      ( (__attribute__((address_space(3))) const char*) p );
}

__device__ __forceinline__ void async_b128_to_lds(unsigned lds_byte,
                                                  unsigned long long gaddr) {
  asm volatile("global_load_async_to_lds_b128 %0, %1, off"
               :: "v"(lds_byte), "v"(gaddr) : "memory");
}

__device__ __forceinline__ void wait_async0() {
  asm volatile("s_wait_asynccnt 0x0" ::: "memory");
}

// ---------------- pre-pass: W1 (f32 [k][n]) -> W1T (bf16 [n][k]) ----------
__global__ __launch_bounds__(256)
void w1_transpose_kernel(const float* __restrict__ W1,
                         unsigned int* __restrict__ W1T32) {
  int idx = blockIdx.x * 256 + threadIdx.x;   // 0..131071 (512x256 k-pairs)
  int n  = idx & (N_ - 1);
  int kp = idx >> 9;                          // 0..255
  float a = W1[(long)(2 * kp)     * N_ + n];
  float c = W1[(long)(2 * kp + 1) * N_ + n];
  W1T32[n * (C_ / 2) + kp] = (unsigned)f2bf(a) | ((unsigned)f2bf(c) << 16);
}

// ---------------- fused GEMM1 + ReLU + GEMM2 + CE + argmax ----------------
__global__ __launch_bounds__(256)
void merger_gemm_kernel(const float* __restrict__ feat,
                        const float* __restrict__ mask,
                        const int*   __restrict__ tgt,
                        const unsigned short* __restrict__ w1t, // bf16 [n][k]
                        const float* __restrict__ b1,
                        const float* __restrict__ W2,
                        const float* __restrict__ b2,
                        float* __restrict__ out_logits,
                        float* __restrict__ ws_acc,   // [0]=loss sum, [1]=count
                        int*   __restrict__ ws_mm)
{
  __shared__ unsigned int sA32[2][64 * LDA32];   // double-buffered A (packed bf16)
  __shared__ unsigned int sB32[N_ * LDB32];      // B tile (packed bf16), async-filled
  __shared__ float sLogits[64 * 3];
  __shared__ float sRed[2];

  const int tid  = threadIdx.x;
  const int lane = tid & 31;
  const int wave = tid >> 5;
  const int mt   = wave & 3;        // 16-row M sub-tile (0..3)
  const int nh   = wave >> 2;       // 256-col N half (0..1)
  const int hiK  = lane >> 4;       // lane half (WMMA layout)

  const int blk = blockIdx.x;
  const int b   = blk >> 8;                 // 256 blocks per batch (HW/64)
  const int t0  = (blk & 255) * 64;

  for (int i = tid; i < 64 * 3; i += 256) sLogits[i] = 0.f;
  if (tid == 0) { sRed[0] = 0.f; sRed[1] = 0.f; }

  v8f acc[16];
  #pragma unroll
  for (int nt = 0; nt < 16; ++nt)
    #pragma unroll
    for (int e = 0; e < 8; ++e) acc[nt][e] = 0.f;

  const int m_loc = mt * 16 + (lane & 15);
  const unsigned sB_base = lds_off(&sB32[0]);
  const unsigned long long w1t_addr = (unsigned long long)w1t;

  // per-thread async-copy coordinates (8 x b128 = 128B of the 32KB tile)
  // q = tid + it*256 ; n = q>>2 ; seg = q&3  (seg = 16B chunk within 64B row)
  // staging helpers ------------------------------------------------------
  auto stageA = [&](int kk, int buf) {
    const long fbase = ((long)b * C_ + kk) * HW_ + t0;
    #pragma unroll
    for (int it = 0; it < 4; ++it) {         // 1024 k-pairs / 256 threads
      int p  = tid + it * 256;
      int m  = p & 63;
      int kp = p >> 6;                        // 0..15
      long gi = fbase + (long)(2 * kp) * HW_ + m;
      long gj = gi + HW_;
      float v0 = feat[gi] * mask[gi];
      float v1 = feat[gj] * mask[gj];
      sA32[buf][m * LDA32 + kp] =
          (unsigned)f2bf(v0) | ((unsigned)f2bf(v1) << 16);
    }
  };
  auto stageB_async = [&](int kk) {
    #pragma unroll
    for (int it = 0; it < 8; ++it) {         // 2048 b128 ops / 256 threads
      int q   = tid + it * 256;
      int n   = q >> 2;
      int seg = q & 3;
      unsigned ldsb = sB_base + (unsigned)(n * (LDB32 * 4) + seg * 16);
      unsigned long long ga =
          w1t_addr + (unsigned long long)n * (C_ * 2) + (unsigned)(kk * 2 + seg * 16);
      async_b128_to_lds(ldsb, ga);
    }
  };

  // prologue: stage step 0
  stageB_async(0);
  stageA(0, 0);
  wait_async0();
  __syncthreads();

  for (int step = 0; step < 16; ++step) {
    const int cur = step & 1;

    // ---- compute from sA32[cur], sB32 ----
    FragAB af;
    #pragma unroll
    for (int g = 0; g < 8; ++g) {
      int kp = g + (hiK << 3);   // A layout (16x32 bf16)
      af.u[g] = sA32[cur][m_loc * LDA32 + kp];
    }
    const int n0 = nh * 256 + (lane & 15);
    FragAB bfr[2];
    #pragma unroll
    for (int g = 0; g < 8; ++g)                        // prefetch nt=0
      bfr[0].u[g] = sB32[n0 * LDB32 + g + (hiK << 3)];
    #pragma unroll
    for (int nt = 0; nt < 16; ++nt) {
      if (nt + 1 < 16) {                               // prefetch nt+1
        int n_loc = n0 + (nt + 1) * 16;
        #pragma unroll
        for (int g = 0; g < 8; ++g)
          bfr[(nt + 1) & 1].u[g] = sB32[n_loc * LDB32 + g + (hiK << 3)];
      }
      acc[nt] = __builtin_amdgcn_wmma_f32_16x16x32_bf16(
          false, af.v, false, bfr[nt & 1].v, (short)0, acc[nt], false, false);
    }
    __syncthreads();                                   // everyone done with sB32

    if (step + 1 < 16) {
      stageB_async((step + 1) * 32);                   // DMA next B tile
      stageA((step + 1) * 32, cur ^ 1);                // overlap VALU A staging
      wait_async0();
      __syncthreads();
    }
  }

  // ---- epilogue: bias + ReLU + fused GEMM2 (N=3) ----
  float b1l[16], w20[16], w21[16], w22[16];
  #pragma unroll
  for (int nt = 0; nt < 16; ++nt) {
    int n = nh * 256 + nt * 16 + (lane & 15);
    b1l[nt] = b1[n];
    w20[nt] = W2[n * 3 + 0];
    w21[nt] = W2[n * 3 + 1];
    w22[nt] = W2[n * 3 + 2];
  }
  #pragma unroll
  for (int r = 0; r < 8; ++r) {
    float s0 = 0.f, s1 = 0.f, s2 = 0.f;
    #pragma unroll
    for (int nt = 0; nt < 16; ++nt) {
      float h = fmaxf(acc[nt][r] + b1l[nt], 0.f);
      s0 += h * w20[nt]; s1 += h * w21[nt]; s2 += h * w22[nt];
    }
    #pragma unroll
    for (int d = 1; d < 16; d <<= 1) {   // butterfly within 16-lane halves
      s0 += __shfl_xor(s0, d);
      s1 += __shfl_xor(s1, d);
      s2 += __shfl_xor(s2, d);
    }
    if ((lane & 15) == 0) {
      int m = mt * 16 + r + (hiK << 3);
      atomicAdd(&sLogits[m * 3 + 0], s0);
      atomicAdd(&sLogits[m * 3 + 1], s1);
      atomicAdd(&sLogits[m * 3 + 2], s2);
    }
  }
  __syncthreads();

  if (tid < 64) {
    int t = t0 + tid;
    float l0 = sLogits[tid * 3 + 0] + b2[0];
    float l1 = sLogits[tid * 3 + 1] + b2[1];
    float l2 = sLogits[tid * 3 + 2] + b2[2];
    long obase = ((long)b * HW_ + t) * 3;
    out_logits[obase + 0] = l0;
    out_logits[obase + 1] = l1;
    out_logits[obase + 2] = l2;
    // cross-entropy (ignore_index = -1)
    int tv = tgt[(long)b * HW_ + t];
    float mx = fmaxf(l0, fmaxf(l1, l2));
    float lse = __logf(__expf(l0 - mx) + __expf(l1 - mx) + __expf(l2 - mx)) + mx;
    if (tv >= 0) {
      float lt = (tv == 0) ? l0 : ((tv == 1) ? l1 : l2);
      atomicAdd(&sRed[0], lse - lt);
      atomicAdd(&sRed[1], 1.f);
    }
    // argmax (first-max) + boundary fixups
    int mm = 0; float best = l0;
    if (l1 > best) { best = l1; mm = 1; }
    if (l2 > best) { mm = 2; }
    int row = t / W_, col = t % W_;
    if (row == 0 && mm == 2) mm = 0;
    if (col == 0 && mm == 1) mm = 0;
    if (row == 0 && col == 0) mm = 0;
    ws_mm[(long)b * HW_ + t] = mm;
  }
  __syncthreads();
  if (tid == 0) {
    atomicAdd(&ws_acc[0], sRed[0]);
    atomicAdd(&ws_acc[1], sRed[1]);
  }
}

__global__ void finalize_loss_kernel(const float* __restrict__ ws_acc,
                                     float* __restrict__ out_loss) {
  out_loss[0] = ws_acc[0] / fmaxf(ws_acc[1], 1.f);
}

// spans[...,0] = mm ; spans[...,2] = colspan (run of 1s starting at j+1)
__global__ void spans_row_kernel(const int* __restrict__ mm,
                                 float* __restrict__ spans) {
  int idx = blockIdx.x * blockDim.x + threadIdx.x;   // b*H + i
  if (idx >= B_ * H_) return;
  const int* row = mm + (long)idx * W_;
  float* sp = spans + (long)idx * W_ * 3;
  int run = 0;
  for (int j = W_ - 1; j >= 0; --j) {
    int v = row[j];
    sp[j * 3 + 0] = (float)v;
    sp[j * 3 + 2] = (float)run;    // run_c[j+1], last col = 0
    run = (v == 1) ? run + 1 : 0;
  }
}

// spans[...,1] = rowspan (run of 2s starting at i+1)
__global__ void spans_col_kernel(const int* __restrict__ mm,
                                 float* __restrict__ spans) {
  int idx = blockIdx.x * blockDim.x + threadIdx.x;   // b*W + j
  if (idx >= B_ * W_) return;
  int b = idx / W_, j = idx % W_;
  int run = 0;
  for (int i = H_ - 1; i >= 0; --i) {
    long off = ((long)b * H_ + i) * W_ + j;
    int v = mm[off];
    spans[off * 3 + 1] = (float)run;   // run_r[i+1], last row = 0
    run = (v == 2) ? run + 1 : 0;
  }
}

extern "C" void kernel_launch(void* const* d_in, const int* in_sizes, int n_in,
                              void* d_out, int out_size, void* d_ws, size_t ws_size,
                              hipStream_t stream) {
  const float* feat = (const float*)d_in[0];
  const float* mask = (const float*)d_in[1];
  const int*   tgt  = (const int*)d_in[2];
  const float* W1   = (const float*)d_in[3];
  const float* b1   = (const float*)d_in[4];
  const float* W2   = (const float*)d_in[5];
  const float* b2   = (const float*)d_in[6];

  float* out        = (float*)d_out;
  float* out_logits = out;                         // 4*128*128*3
  float* out_loss   = out + (long)B_ * HW_ * 3;    // 1
  float* out_spans  = out_loss + 1;                // 4*128*128*3

  // workspace layout: [0,16) loss accumulators; [256, 256+512K) W1T bf16;
  // [524544, 524544+256K) mm labels.  (~768 KB total)
  float*          ws_acc = (float*)d_ws;
  unsigned int*   w1t32  = (unsigned int*)((char*)d_ws + 256);
  int*            ws_mm  = (int*)((char*)d_ws + 256 + (size_t)C_ * N_ * 2);

  hipMemsetAsync(d_ws, 0, 16, stream);             // zero loss accumulators

  w1_transpose_kernel<<<(C_ * N_ / 2) / 256, 256, 0, stream>>>(W1, w1t32);
  merger_gemm_kernel<<<B_ * HW_ / 64, 256, 0, stream>>>(
      feat, mask, tgt, (const unsigned short*)w1t32, b1, W2, b2,
      out_logits, ws_acc, ws_mm);
  finalize_loss_kernel<<<1, 1, 0, stream>>>(ws_acc, out_loss);
  spans_row_kernel<<<(B_ * H_ + 255) / 256, 256, 0, stream>>>(ws_mm, out_spans);
  spans_col_kernel<<<(B_ * W_ + 255) / 256, 256, 0, stream>>>(ws_mm, out_spans);
}
